// AttentionBlock_51651276702104
// MI455X (gfx1250) — matrix operand order repaired and checked by reading, not compiled, not run on hardware
//
#include <hip/hip_runtime.h>
#include <math.h>
#include <stdint.h>

// ---------------------------------------------------------------------------
// AttentionBlock for MI455X (gfx1250): GroupNorm -> QKV -> flash attention ->
// projection + residual.  GEMMs on v_wmma_f32_16x16x32_bf16 (wave32); K/V
// staging in the attention kernel uses async global->LDS DMA (ASYNCcnt) with
// double-buffered K tiles so transfers overlap WMMA compute.
// ---------------------------------------------------------------------------

typedef __bf16 bf16_t;
typedef __bf16 v16bf __attribute__((ext_vector_type(16)));
typedef __bf16 v8bf  __attribute__((ext_vector_type(8)));
typedef float  v8f   __attribute__((ext_vector_type(8)));
typedef float  v4f   __attribute__((ext_vector_type(4)));

static constexpr int BATCH = 8;
static constexpr int NTOK  = 4096;          // 64*64 tokens per batch
static constexpr int CCH   = 256;           // channels
static constexpr int MROW  = BATCH * NTOK;  // 32768 flat rows

// ---- WMMA fragment loaders -------------------------------------------------
// A (16x32 bf16): lane<16 holds row=lane&15, K = {0..7, 16..23};
//                 lane>=16 holds K = {8..15, 24..31}.  Two 16B loads.
__device__ __forceinline__ v16bf frag_a_load(const bf16_t* lanePtr, int lane) {
  const int koff = (lane < 16) ? 0 : 8;
  v8bf lo = *(const v8bf*)(lanePtr + koff);
  v8bf hi = *(const v8bf*)(lanePtr + koff + 16);
  v16bf r;
#pragma unroll
  for (int i = 0; i < 8; ++i) { r[i] = lo[i]; r[8 + i] = hi[i]; }
  return r;
}
// B (32x16 bf16): lane<16 holds col=lane&15, K = {0..7, 16..23};
// lane>=16 K = {8..15, 24..31} (requires K-major packed operand).
__device__ __forceinline__ v16bf frag_b_load(const bf16_t* lanePtr, int lane) {
  const int koff = (lane < 16) ? 0 : 8;
  v8bf lo = *(const v8bf*)(lanePtr + koff);
  v8bf hi = *(const v8bf*)(lanePtr + koff + 16);
  v16bf r;
#pragma unroll
  for (int i = 0; i < 8; ++i) { r[i] = lo[i]; r[8 + i] = hi[i]; }
  return r;
}

__device__ __forceinline__ v8f wmma_bf16(v16bf a, v16bf b, v8f c) {
  return __builtin_amdgcn_wmma_f32_16x16x32_bf16(false, a, false, b, (short)0, c,
                                                 false, false);
}

// ---- async global -> LDS DMA (gfx1250, tracked by ASYNCcnt) ----------------
// Per-lane: 16B from global address -> 16B at 32-bit LDS address.
__device__ __forceinline__ void async_b128(unsigned lds_addr, uint64_t gaddr) {
  asm volatile("global_load_async_to_lds_b128 %0, %1, off"
               :: "v"(lds_addr), "v"(gaddr)
               : "memory");
}

#define V8F_ZERO {0.f,0.f,0.f,0.f,0.f,0.f,0.f,0.f}

// ---- GroupNorm(1 group): per-batch mean / rstd -----------------------------
__global__ __launch_bounds__(256) void gn_stats_kernel(const float* __restrict__ x,
                                                       float* __restrict__ stats) {
  __shared__ float s1[256];
  __shared__ float s2[256];
  const int b = blockIdx.x;
  const v4f* xp = (const v4f*)(x + (size_t)b * (NTOK * CCH));
  float sum = 0.f, sq = 0.f;
  for (int i = threadIdx.x; i < (NTOK * CCH) / 4; i += 256) {
    v4f v = xp[i];
    sum += v[0] + v[1] + v[2] + v[3];
    sq  += v[0]*v[0] + v[1]*v[1] + v[2]*v[2] + v[3]*v[3];
  }
  s1[threadIdx.x] = sum; s2[threadIdx.x] = sq;
  __syncthreads();
  for (int st = 128; st > 0; st >>= 1) {
    if (threadIdx.x < st) { s1[threadIdx.x] += s1[threadIdx.x + st];
                            s2[threadIdx.x] += s2[threadIdx.x + st]; }
    __syncthreads();
  }
  if (threadIdx.x == 0) {
    const float inv = 1.f / (float)(NTOK * CCH);
    float mean = s1[0] * inv;
    float var  = s2[0] * inv - mean * mean;
    stats[b * 2 + 0] = mean;
    stats[b * 2 + 1] = rsqrtf(var + 1e-3f);
  }
}

// ---- GroupNorm apply -> h (bf16) -------------------------------------------
__global__ __launch_bounds__(256) void gn_apply_kernel(const float* __restrict__ x,
                                                       const float* __restrict__ gamma,
                                                       const float* __restrict__ beta,
                                                       const float* __restrict__ stats,
                                                       bf16_t* __restrict__ h) {
  const size_t i = ((size_t)blockIdx.x * 256 + threadIdx.x) * 4;
  const int b = (int)(i >> 20);             // NTOK*CCH = 2^20 per batch
  const int c = (int)(i & (CCH - 1));
  const float mean = stats[b * 2 + 0];
  const float rstd = stats[b * 2 + 1];
  v4f v = *(const v4f*)(x + i);
#pragma unroll
  for (int u = 0; u < 4; ++u)
    h[i + u] = (bf16_t)((v[u] - mean) * rstd * gamma[c + u] + beta[c + u]);
}

// ---- Weight repack: f32 [k][n] -> bf16 [n][k] (K-major per output column) --
__global__ __launch_bounds__(256) void repack_kernel(const float* __restrict__ w,
                                                     bf16_t* __restrict__ wt) {
  const int n = blockIdx.x, k = threadIdx.x;
  wt[n * CCH + k] = (bf16_t)w[k * CCH + n];
}

// ---- Fused QKV GEMM: q,k row-major bf16; v stored transposed vt[b][c][tok] -
__global__ __launch_bounds__(256) void qkv_kernel(
    const bf16_t* __restrict__ h,
    const bf16_t* __restrict__ wqt, const bf16_t* __restrict__ wkt,
    const bf16_t* __restrict__ wvt,
    const float* __restrict__ bq, const float* __restrict__ bk,
    const float* __restrict__ bv,
    bf16_t* __restrict__ q, bf16_t* __restrict__ k, bf16_t* __restrict__ vt) {
  const int lane = threadIdx.x & 31;
  const int wave = threadIdx.x >> 5;
  const size_t m0 = (size_t)blockIdx.x * 128 + wave * 16;
  const int n0 = blockIdx.y * 32;
  const int cl = lane & 15;

  v8f aq0 = V8F_ZERO, aq1 = V8F_ZERO, ak0 = V8F_ZERO,
      ak1 = V8F_ZERO, av0 = V8F_ZERO, av1 = V8F_ZERO;

  const bf16_t* hp = h + (m0 + cl) * CCH;
#pragma unroll
  for (int kc = 0; kc < 8; ++kc) {
    v16bf af = frag_a_load(hp + kc * 32, lane);
    v16bf q0 = frag_b_load(wqt + (size_t)(n0 + cl)      * CCH + kc * 32, lane);
    v16bf q1 = frag_b_load(wqt + (size_t)(n0 + 16 + cl) * CCH + kc * 32, lane);
    v16bf k0 = frag_b_load(wkt + (size_t)(n0 + cl)      * CCH + kc * 32, lane);
    v16bf k1 = frag_b_load(wkt + (size_t)(n0 + 16 + cl) * CCH + kc * 32, lane);
    v16bf v0 = frag_b_load(wvt + (size_t)(n0 + cl)      * CCH + kc * 32, lane);
    v16bf v1 = frag_b_load(wvt + (size_t)(n0 + 16 + cl) * CCH + kc * 32, lane);
    aq0 = wmma_bf16(af, q0, aq0);  aq1 = wmma_bf16(af, q1, aq1);
    ak0 = wmma_bf16(af, k0, ak0);  ak1 = wmma_bf16(af, k1, ak1);
    av0 = wmma_bf16(af, v0, av0);  av1 = wmma_bf16(af, v1, av1);
  }

  const int rbase = (lane < 16) ? 0 : 8;
  const int c0 = n0 + cl, c1 = n0 + 16 + cl;
  const float bq0 = bq[c0], bq1 = bq[c1];
  const float bk0 = bk[c0], bk1 = bk[c1];
  const float bv0 = bv[c0], bv1 = bv[c1];
#pragma unroll
  for (int j = 0; j < 8; ++j) {
    const size_t row = m0 + rbase + j;
    q[row * CCH + c0] = (bf16_t)(aq0[j] + bq0);
    q[row * CCH + c1] = (bf16_t)(aq1[j] + bq1);
    k[row * CCH + c0] = (bf16_t)(ak0[j] + bk0);
    k[row * CCH + c1] = (bf16_t)(ak1[j] + bk1);
    const size_t bb = row >> 12;          // row / NTOK
    const size_t t  = row & (NTOK - 1);
    vt[(bb * CCH + c0) * NTOK + t] = (bf16_t)(av0[j] + bv0);
    vt[(bb * CCH + c1) * NTOK + t] = (bf16_t)(av1[j] + bv1);
  }
}

// ---- Flash attention -------------------------------------------------------
// Each 256-thread block owns 128 query rows (16 per wave).  Per 32-key chunk:
// async-DMA V[i] and K[i+1] into LDS (double-buffered K) while computing
// S[i] = Q K^T from the resident K buffer; online softmax; P round-trips
// through per-wave LDS scratch to A-fragment layout; O += P V.
__global__ __launch_bounds__(256) void attn_kernel(const bf16_t* __restrict__ q,
                                                   const bf16_t* __restrict__ k,
                                                   const bf16_t* __restrict__ vt,
                                                   bf16_t* __restrict__ o) {
  __shared__ __align__(16) bf16_t Kbuf[2][32 * CCH]; // 2 x 16 KB, double-buffer
  __shared__ __align__(16) bf16_t Vt[CCH * 32];      // 16 KB: 256 ch x 32 keys
  __shared__ __align__(16) bf16_t Pt[8 * 16 * 32];   //  8 KB: per-wave P scratch

  const int lane = threadIdx.x & 31;
  const int wave = threadIdx.x >> 5;
  const int cl = lane & 15;
  const int b = blockIdx.y;
  const size_t qrow0 = (size_t)b * NTOK + blockIdx.x * 128 + wave * 16;

  const uint64_t kbase = (uint64_t)(uintptr_t)(k + (size_t)b * NTOK * CCH);
  const uint64_t vbase =
      (uint64_t)(uintptr_t)(vt + ((size_t)b * CCH + threadIdx.x) * NTOK);

  // preload Q A-fragments: 8 k-steps over C=256
  v16bf qf[8];
  {
    const bf16_t* qp = q + (qrow0 + cl) * CCH;
#pragma unroll
    for (int kc = 0; kc < 8; ++kc) qf[kc] = frag_a_load(qp + kc * 32, lane);
  }

  v8f acc[16];
#pragma unroll
  for (int t = 0; t < 16; ++t) acc[t] = (v8f)V8F_ZERO;
  float mrow[8], lrow[8];
#pragma unroll
  for (int j = 0; j < 8; ++j) { mrow[j] = -INFINITY; lrow[j] = 0.f; }

  bf16_t* pw = Pt + wave * (16 * 32);

  // prologue: async-load K chunk 0 into Kbuf[0] (4 x b128 per wave)
  {
    const unsigned d0 = (unsigned)(uintptr_t)(&Kbuf[0][0]);
#pragma unroll
    for (int i = 0; i < 4; ++i) {
      const unsigned idx = threadIdx.x + i * 256;   // 16B-chunk index
      async_b128(d0 + idx * 16, kbase + (uint64_t)idx * 16);
    }
    asm volatile("s_wait_asynccnt 0x0" ::: "memory");
  }

  for (int it = 0; it < NTOK / 32; ++it) {
    const int kb = it * 32;
    // publishes K[it] (each wave waited its ASYNCcnt before arriving here);
    // also: every wave is done reading Vt (PV[it-1]) and Kbuf[(it+1)&1].
    __syncthreads();

    // async-issue V[it] (4 x b128/wave) then K[it+1] (4 x b128/wave)
    {
      const unsigned vd = (unsigned)(uintptr_t)(&Vt[threadIdx.x * 32]);
      const uint64_t vs = vbase + (uint64_t)kb * 2;   // 64B row slice
#pragma unroll
      for (int i = 0; i < 4; ++i) async_b128(vd + i * 16, vs + i * 16);
    }
    if (it + 1 < NTOK / 32) {
      const unsigned kd = (unsigned)(uintptr_t)(&Kbuf[(it + 1) & 1][0]);
      const uint64_t ks = kbase + (uint64_t)(kb + 32) * CCH * 2;
#pragma unroll
      for (int i = 0; i < 4; ++i) {
        const unsigned idx = threadIdx.x + i * 256;
        async_b128(kd + idx * 16, ks + (uint64_t)idx * 16);
      }
    }

    // S = Q K^T for this 16x32 key slab (overlaps the DMAs above)
    const bf16_t* Kt = Kbuf[it & 1];
    v8f s0 = V8F_ZERO, s1 = V8F_ZERO;
#pragma unroll
    for (int kc = 0; kc < 8; ++kc) {
      const bf16_t* kp = Kt + kc * 32;
      v16bf b0 = frag_b_load(kp + (size_t)cl * CCH, lane);         // keys 0..15
      v16bf b1 = frag_b_load(kp + (size_t)(cl + 16) * CCH, lane);  // keys 16..31
      s0 = wmma_bf16(qf[kc], b0, s0);
      s1 = wmma_bf16(qf[kc], b1, s1);
    }

    // online softmax (rows live across 16-lane halves)
    const float sc = 0.0625f;  // C^-0.5
    float p0[8], p1[8];
#pragma unroll
    for (int j = 0; j < 8; ++j) {
      float a = s0[j] * sc, c2 = s1[j] * sc;
      float mx = fmaxf(a, c2);
#pragma unroll
      for (int d = 1; d < 16; d <<= 1) mx = fmaxf(mx, __shfl_xor(mx, d, 32));
      float mnew = fmaxf(mrow[j], mx);
      float f = __expf(mrow[j] - mnew);
      float e0 = __expf(a - mnew), e1 = __expf(c2 - mnew);
      float rs = e0 + e1;
#pragma unroll
      for (int d = 1; d < 16; d <<= 1) rs += __shfl_xor(rs, d, 32);
      lrow[j] = lrow[j] * f + rs;
      mrow[j] = mnew;
#pragma unroll
      for (int t = 0; t < 16; ++t) acc[t][j] *= f;
      p0[j] = e0; p1[j] = e1;
    }

    // my V slice done (K[it+1] may still be in flight: ASYNCcnt <= 4 keeps
    // exactly the 4 younger K ops outstanding thanks to in-order retirement)
    if (it + 1 < NTOK / 32) {
      asm volatile("s_wait_asynccnt 0x4" ::: "memory");
    } else {
      asm volatile("s_wait_asynccnt 0x0" ::: "memory");
    }
    __syncthreads();  // all waves' V[it] slices now visible

    // C/D layout -> A layout via per-wave LDS round trip
    {
      const int rbase = (lane < 16) ? 0 : 8;
#pragma unroll
      for (int j = 0; j < 8; ++j) {
        pw[(rbase + j) * 32 + cl]      = (bf16_t)p0[j];
        pw[(rbase + j) * 32 + cl + 16] = (bf16_t)p1[j];
      }
    }
    v16bf pf = frag_a_load(pw + (size_t)cl * 32, lane);

    // O += P @ V over 16 channel tiles
#pragma unroll
    for (int t = 0; t < 16; ++t) {
      v16bf vf = frag_b_load(Vt + (size_t)(t * 16 + cl) * 32, lane);
      acc[t] = wmma_bf16(pf, vf, acc[t]);
    }

    // my K[it+1] slice done before the next top barrier publishes it
    asm volatile("s_wait_asynccnt 0x0" ::: "memory");
  }

  // epilogue: O /= l, emit bf16
  {
    const int rbase = (lane < 16) ? 0 : 8;
#pragma unroll
    for (int j = 0; j < 8; ++j) {
      const float inv = 1.f / lrow[j];
      const size_t row = qrow0 + rbase + j;
#pragma unroll
      for (int t = 0; t < 16; ++t)
        o[row * CCH + t * 16 + cl] = (bf16_t)(acc[t][j] * inv);
    }
  }
}

// ---- Output projection + residual (fp32 out) -------------------------------
__global__ __launch_bounds__(256) void proj_kernel(const bf16_t* __restrict__ o,
                                                   const bf16_t* __restrict__ wpt,
                                                   const float* __restrict__ bp,
                                                   const float* __restrict__ x,
                                                   float* __restrict__ out) {
  const int lane = threadIdx.x & 31;
  const int wave = threadIdx.x >> 5;
  const size_t m0 = (size_t)blockIdx.x * 128 + wave * 16;
  const int n0 = blockIdx.y * 32;
  const int cl = lane & 15;

  v8f a0 = V8F_ZERO, a1 = V8F_ZERO;
  const bf16_t* op = o + (m0 + cl) * CCH;
#pragma unroll
  for (int kc = 0; kc < 8; ++kc) {
    v16bf af = frag_a_load(op + kc * 32, lane);
    v16bf b0 = frag_b_load(wpt + (size_t)(n0 + cl)      * CCH + kc * 32, lane);
    v16bf b1 = frag_b_load(wpt + (size_t)(n0 + 16 + cl) * CCH + kc * 32, lane);
    a0 = wmma_bf16(af, b0, a0);
    a1 = wmma_bf16(af, b1, a1);
  }
  const int rbase = (lane < 16) ? 0 : 8;
  const int c0 = n0 + cl, c1 = n0 + 16 + cl;
  const float bp0 = bp[c0], bp1 = bp[c1];
#pragma unroll
  for (int j = 0; j < 8; ++j) {
    const size_t row = m0 + rbase + j;
    out[row * CCH + c0] = x[row * CCH + c0] + a0[j] + bp0;
    out[row * CCH + c1] = x[row * CCH + c1] + a1[j] + bp1;
  }
}

// ---------------------------------------------------------------------------
extern "C" void kernel_launch(void* const* d_in, const int* in_sizes, int n_in,
                              void* d_out, int out_size, void* d_ws, size_t ws_size,
                              hipStream_t stream) {
  const float* x     = (const float*)d_in[0];
  const float* gamma = (const float*)d_in[1];
  const float* beta  = (const float*)d_in[2];
  const float* wq    = (const float*)d_in[3];
  const float* bq    = (const float*)d_in[4];
  const float* wk    = (const float*)d_in[5];
  const float* bk    = (const float*)d_in[6];
  const float* wv    = (const float*)d_in[7];
  const float* bv    = (const float*)d_in[8];
  const float* wp    = (const float*)d_in[9];
  const float* bp    = (const float*)d_in[10];
  float* out = (float*)d_out;

  // workspace carve-up (~84.4 MB total)
  char* w = (char*)d_ws;
  float*  stats = (float*)w;                       // 64 B (padded to 256)
  bf16_t* wqt = (bf16_t*)(w + 256);                // 4 x 128 KB packed weights
  bf16_t* wkt = wqt + CCH * CCH;
  bf16_t* wvt = wkt + CCH * CCH;
  bf16_t* wpt = wvt + CCH * CCH;
  bf16_t* h   = wpt + CCH * CCH;                   // 5 x 16 MB activations
  bf16_t* q   = h  + (size_t)MROW * CCH;
  bf16_t* k   = q  + (size_t)MROW * CCH;
  bf16_t* vt  = k  + (size_t)MROW * CCH;
  bf16_t* o   = vt + (size_t)MROW * CCH;

  gn_stats_kernel<<<BATCH, 256, 0, stream>>>(x, stats);
  gn_apply_kernel<<<(MROW * CCH) / (256 * 4), 256, 0, stream>>>(x, gamma, beta,
                                                                stats, h);
  repack_kernel<<<CCH, CCH, 0, stream>>>(wq, wqt);
  repack_kernel<<<CCH, CCH, 0, stream>>>(wk, wkt);
  repack_kernel<<<CCH, CCH, 0, stream>>>(wv, wvt);
  repack_kernel<<<CCH, CCH, 0, stream>>>(wp, wpt);
  qkv_kernel<<<dim3(MROW / 128, CCH / 32), 256, 0, stream>>>(
      h, wqt, wkt, wvt, bq, bk, bv, q, k, vt);
  attn_kernel<<<dim3(NTOK / 128, BATCH), 256, 0, stream>>>(q, k, vt, o);
  proj_kernel<<<dim3(MROW / 128, CCH / 32), 256, 0, stream>>>(o, wpt, bp, x, out);
}
